// HyperMixerLayer_8435315769841
// MI455X (gfx1250) — matrix operand order repaired and checked by reading, not compiled, not run on hardware
//
#include <hip/hip_runtime.h>
#include <hip/hip_bf16.h>

#define TT 75000
#define DD 256
#define BG 50

typedef _Float16 half_t;
typedef __attribute__((ext_vector_type(16))) _Float16 v16h;
typedef __attribute__((ext_vector_type(8)))  _Float16 v8h;
typedef __attribute__((ext_vector_type(8)))  float    v8f;

__device__ __forceinline__ float gelu_f(float x) {
  return 0.5f * x * (1.0f + erff(x * 0.7071067811865476f));
}

__device__ __forceinline__ v8f wmma16(v16h a, v16h b, v8f c) {
  return __builtin_amdgcn_wmma_f32_16x16x32_f16(false, a, false, b, (short)0, c, false, false);
}

// A fragment (16x32, f16) from row-major A[M][K], stride s halfs. Rows clamped to mmax.
// Lane layout (ISA 7.12.2): lane L -> M = m0 + L%16; halves 0..7 = K[kb..kb+7],
// halves 8..15 = K[kb+16..kb+23], kb = k0 + (L/16)*8.
__device__ __forceinline__ v16h load_a_rm(const half_t* __restrict__ A, int s, int m0, int k0, int mmax) {
  int lane = threadIdx.x & 31;
  int m = m0 + (lane & 15); if (m > mmax) m = mmax;
  int kb = k0 + ((lane >> 4) << 3);
  const half_t* p = A + m * s + kb;
  v8h lo = *(const v8h*)p;
  v8h hi = *(const v8h*)(p + 16);
  v16h r;
#pragma unroll
  for (int i = 0; i < 8; ++i) { r[i] = lo[i]; r[i + 8] = hi[i]; }
  return r;
}

// B fragment (32x16, f16) from the K-transposed matrix Bt[N][K], stride s halfs.
// Lane L -> N = n0 + L%16; halves 0..7 = K[kb..kb+7], halves 8..15 = K[kb+16..kb+23], kb = k0 + (L/16)*8.
__device__ __forceinline__ v16h load_bt(const half_t* __restrict__ Bt, int s, int n0, int k0) {
  int lane = threadIdx.x & 31;
  int n = n0 + (lane & 15);
  int kb = k0 + ((lane >> 4) << 3);
  const half_t* p = Bt + n * s + kb;
  v8h lo = *(const v8h*)p;
  v8h hi = *(const v8h*)(p + 16);
  v16h r;
#pragma unroll
  for (int i = 0; i < 8; ++i) { r[i] = lo[i]; r[i + 8] = hi[i]; }
  return r;
}

// Masked (segment edge) variants: zero outside [kst, ken). Scalar u16 loads; used only
// for the first/last K-chunk of a graph segment (interior chunks are 32-aligned).
__device__ __forceinline__ v16h load_a_seg(const half_t* __restrict__ A, int s, int m0,
                                           int k0, int kst, int ken) {
  int lane = threadIdx.x & 31;
  int m = m0 + (lane & 15);
  int kb = k0 + ((lane >> 4) << 3);
  const half_t* p = A + m * s;
  v16h r;
#pragma unroll
  for (int i = 0; i < 8; ++i) {
    int ka = kb + i, kc = kb + 16 + i;
    r[i]     = (ka >= kst && ka < ken) ? p[ka] : (half_t)0.0f;
    r[i + 8] = (kc >= kst && kc < ken) ? p[kc] : (half_t)0.0f;
  }
  return r;
}
__device__ __forceinline__ v16h load_b_seg(const half_t* __restrict__ Bt, int s, int n0,
                                           int k0, int kst, int ken) {
  int lane = threadIdx.x & 31;
  int n = n0 + (lane & 15);
  int kb = k0 + ((lane >> 4) << 3);
  const half_t* p = Bt + n * s;
  v16h r;
#pragma unroll
  for (int i = 0; i < 16; ++i) {
    int k = kb + i + ((i >> 3) << 3);
    r[i] = (k >= kst && k < ken) ? p[k] : (half_t)0.0f;
  }
  return r;
}

// ---------------- segment bookkeeping ----------------
__global__ void k_init_seg(int* starts, int* ends) {
  int g = threadIdx.x;
  if (g < BG) { starts[g] = TT; ends[g] = 0; }
}
__global__ void k_seg(const int* __restrict__ batch, int* starts, int* ends) {
  int i = blockIdx.x * blockDim.x + threadIdx.x;
  if (i >= TT) return;
  int g = batch[i];
  atomicMin(&starts[g], i);
  atomicMax(&ends[g], i + 1);
}
__global__ void k_pos(const int* __restrict__ batch, const int* __restrict__ starts, int* __restrict__ pos) {
  int i = blockIdx.x * blockDim.x + threadIdx.x;
  if (i >= TT) return;
  pos[i] = i - starts[batch[i]];
}

// ---------------- layernorm (one wave per row) ----------------
__device__ __forceinline__ void ln_row(const float* __restrict__ src, const float* __restrict__ gam,
                                       const float* __restrict__ bet, int row, int lane, float* y) {
  const float4* r4 = (const float4*)(src + row * DD) + lane * 2;
  float4 a0 = r4[0], a1 = r4[1];
  float v[8] = {a0.x, a0.y, a0.z, a0.w, a1.x, a1.y, a1.z, a1.w};
  float s1 = 0.f, s2 = 0.f;
#pragma unroll
  for (int i = 0; i < 8; ++i) { s1 += v[i]; s2 += v[i] * v[i]; }
#pragma unroll
  for (int off = 16; off; off >>= 1) { s1 += __shfl_xor(s1, off, 32); s2 += __shfl_xor(s2, off, 32); }
  float mu  = s1 * (1.0f / DD);
  float var = s2 * (1.0f / DD) - mu * mu;
  float inv = rsqrtf(var + 1e-5f);
#pragma unroll
  for (int i = 0; i < 8; ++i) { int c = lane * 8 + i; y[i] = (v[i] - mu) * inv * gam[c] + bet[c]; }
}

// LN1: x0 = LN(inp) [f32], x0t = x0^T [f16, D x T], xpeh = f16(x0 + PE[pos]) [T x D]
__global__ void k_ln1(const float* __restrict__ inp, const float* __restrict__ gam,
                      const float* __restrict__ bet, const int* __restrict__ pos,
                      float* __restrict__ x0, half_t* __restrict__ x0t, half_t* __restrict__ xpeh) {
  int row = blockIdx.x * 8 + (threadIdx.x >> 5);
  int lane = threadIdx.x & 31;
  if (row >= TT) return;
  float y[8];
  ln_row(inp, gam, bet, row, lane, y);
  int p = pos[row];
#pragma unroll
  for (int i = 0; i < 8; ++i) {
    int c = lane * 8 + i;
    x0[row * DD + c] = y[i];
    x0t[c * TT + row] = (half_t)y[i];
    float ang = (float)p * expf((float)(c & ~1) * -0.035977889f); // -ln(1e4)/256
    float pe = (c & 1) ? cosf(ang) : sinf(ang);
    xpeh[row * DD + c] = (half_t)(y[i] + pe);
  }
}

// LN2: x2 = LN(x1); stage x2^T, gathered w2^T and gathered w1 (by pos) as f16
__global__ void k_ln2(const float* __restrict__ x1, const float* __restrict__ gam,
                      const float* __restrict__ bet, const int* __restrict__ pos,
                      const float* __restrict__ w1, const float* __restrict__ w2,
                      half_t* __restrict__ x2t, half_t* __restrict__ w2gt, half_t* __restrict__ w1g) {
  int row = blockIdx.x * 8 + (threadIdx.x >> 5);
  int lane = threadIdx.x & 31;
  if (row >= TT) return;
  float y[8];
  ln_row(x1, gam, bet, row, lane, y);
  int p = pos[row];
#pragma unroll
  for (int i = 0; i < 8; ++i) {
    int c = lane * 8 + i;
    x2t[c * TT + row]  = (half_t)y[i];
    w2gt[c * TT + row] = (half_t)w2[p * DD + c];
    w1g[row * DD + c]  = (half_t)w1[p * DD + c];
  }
}

__global__ void k_h16(const float* __restrict__ src, half_t* __restrict__ dst, int n) {
  int i = blockIdx.x * blockDim.x + threadIdx.x;
  if (i < n) dst[i] = (half_t)src[i];
}

// W = gelu(Xpe @ mlp_w^T + b): [T x 256] x [256 x 256]. 32x32 tile per wave,
// block = 8 waves = 32 rows x 256 cols. Stores W row-major and transposed (f16).
__global__ void k_gemm_w(const half_t* __restrict__ xpeh, const half_t* __restrict__ mw,
                         const float* __restrict__ mb, half_t* __restrict__ wh, half_t* __restrict__ wht) {
  int wave = threadIdx.x >> 5, lane = threadIdx.x & 31;
  int m0 = blockIdx.x * 32;
  int n0 = wave * 32;
  v8f a00 = {}, a01 = {}, a10 = {}, a11 = {};
#pragma unroll
  for (int k0 = 0; k0 < DD; k0 += 32) {
    v16h fa0 = load_a_rm(xpeh, DD, m0,      k0, TT - 1);
    v16h fa1 = load_a_rm(xpeh, DD, m0 + 16, k0, TT - 1);
    v16h fb0 = load_bt(mw, DD, n0,      k0);
    v16h fb1 = load_bt(mw, DD, n0 + 16, k0);
    a00 = wmma16(fa0, fb0, a00);
    a01 = wmma16(fa0, fb1, a01);
    a10 = wmma16(fa1, fb0, a10);
    a11 = wmma16(fa1, fb1, a11);
  }
  int halfsel = (lane >> 4) << 3;
  int nlo = n0 + (lane & 15), nhi = nlo + 16;
  float blo = mb[nlo], bhi = mb[nhi];
#pragma unroll
  for (int v = 0; v < 8; ++v) {
    int mrA = m0 + v + halfsel;
    int mrB = mrA + 16;
    if (mrA < TT) {
      float y0 = gelu_f(a00[v] + blo); wh[mrA * DD + nlo] = (half_t)y0; wht[nlo * TT + mrA] = (half_t)y0;
      float y1 = gelu_f(a01[v] + bhi); wh[mrA * DD + nhi] = (half_t)y1; wht[nhi * TT + mrA] = (half_t)y1;
    }
    if (mrB < TT) {
      float y2 = gelu_f(a10[v] + blo); wh[mrB * DD + nlo] = (half_t)y2; wht[nlo * TT + mrB] = (half_t)y2;
      float y3 = gelu_f(a11[v] + bhi); wh[mrB * DD + nhi] = (half_t)y3; wht[nhi * TT + mrB] = (half_t)y3;
    }
  }
}

// Per-graph outer-product GEMM: H[g][e][d] = gelu( sum_{r in seg(g)} At[d][r] * Bt[e][r] ).
// 32x32 tile per wave; K-loop anchored at a 32-aligned offset so interior chunks use
// vectorized loads; only edge chunks take the masked scalar path (wave-uniform branch).
__global__ void k_gemm_outer(const half_t* __restrict__ At, const half_t* __restrict__ Bt,
                             const int* __restrict__ starts, const int* __restrict__ ends,
                             half_t* __restrict__ Ht) {
  int wave = threadIdx.x >> 5, lane = threadIdx.x & 31;
  int g = blockIdx.x >> 3;
  int tile = (blockIdx.x & 7) * 8 + wave;          // 64 tiles of 32x32 over 256x256
  int m0 = (tile >> 3) * 32, n0 = (tile & 7) * 32;
  int st = starts[g], en = ends[g];
  v8f a00 = {}, a01 = {}, a10 = {}, a11 = {};
  for (int k0 = (st & ~31); k0 < en; k0 += 32) {
    v16h fa0, fa1, fb0, fb1;
    if (k0 >= st && k0 + 32 <= en) {               // uniform: EXEC untouched
      fa0 = load_a_rm(At, TT, m0,      k0, DD - 1);
      fa1 = load_a_rm(At, TT, m0 + 16, k0, DD - 1);
      fb0 = load_bt(Bt, TT, n0,      k0);
      fb1 = load_bt(Bt, TT, n0 + 16, k0);
    } else {
      fa0 = load_a_seg(At, TT, m0,      k0, st, en);
      fa1 = load_a_seg(At, TT, m0 + 16, k0, st, en);
      fb0 = load_b_seg(Bt, TT, n0,      k0, st, en);
      fb1 = load_b_seg(Bt, TT, n0 + 16, k0, st, en);
    }
    a00 = wmma16(fa0, fb0, a00);
    a01 = wmma16(fa0, fb1, a01);
    a10 = wmma16(fa1, fb0, a10);
    a11 = wmma16(fa1, fb1, a11);
  }
  half_t* H = Ht + g * (DD * DD);
  int halfsel = (lane >> 4) << 3;
  int nlo = n0 + (lane & 15), nhi = nlo + 16;
#pragma unroll
  for (int v = 0; v < 8; ++v) {
    int mrA = m0 + v + halfsel;
    int mrB = mrA + 16;
    H[nlo * DD + mrA] = (half_t)gelu_f(a00[v]);
    H[nhi * DD + mrA] = (half_t)gelu_f(a01[v]);
    H[nlo * DD + mrB] = (half_t)gelu_f(a10[v]);
    H[nhi * DD + mrB] = (half_t)gelu_f(a11[v]);
  }
}

// Per-graph row GEMM + residual: out[r][e] = sum_d A[r][d]*Ht[g][e][d] + resid[r][e].
// 32x32 tile per wave; block = 32 rows x 256 cols.
__global__ void k_gemm_row(const half_t* __restrict__ A, const half_t* __restrict__ Ht,
                           const int* __restrict__ starts, const int* __restrict__ ends,
                           const float* __restrict__ resid, float* __restrict__ out) {
  int wave = threadIdx.x >> 5, lane = threadIdx.x & 31;
  int g = blockIdx.x / 63;
  int rt = blockIdx.x % 63;
  int st = starts[g], en = ends[g];
  int m0 = st + rt * 32;
  if (m0 >= en) return;  // block-uniform exit
  int n0 = wave * 32;
  const half_t* Bt = Ht + g * (DD * DD);
  v8f a00 = {}, a01 = {}, a10 = {}, a11 = {};
#pragma unroll
  for (int k0 = 0; k0 < DD; k0 += 32) {
    v16h fa0 = load_a_rm(A, DD, m0,      k0, en - 1);
    v16h fa1 = load_a_rm(A, DD, m0 + 16, k0, en - 1);
    v16h fb0 = load_bt(Bt, DD, n0,      k0);
    v16h fb1 = load_bt(Bt, DD, n0 + 16, k0);
    a00 = wmma16(fa0, fb0, a00);
    a01 = wmma16(fa0, fb1, a01);
    a10 = wmma16(fa1, fb0, a10);
    a11 = wmma16(fa1, fb1, a11);
  }
  int halfsel = (lane >> 4) << 3;
  int nlo = n0 + (lane & 15), nhi = nlo + 16;
#pragma unroll
  for (int v = 0; v < 8; ++v) {
    int mrA = m0 + v + halfsel;
    int mrB = mrA + 16;
    if (mrA < en) {
      out[mrA * DD + nlo] = a00[v] + resid[mrA * DD + nlo];
      out[mrA * DD + nhi] = a01[v] + resid[mrA * DD + nhi];
    }
    if (mrB < en) {
      out[mrB * DD + nlo] = a10[v] + resid[mrB * DD + nlo];
      out[mrB * DD + nhi] = a11[v] + resid[mrB * DD + nhi];
    }
  }
}

extern "C" void kernel_launch(void* const* d_in, const int* in_sizes, int n_in,
                              void* d_out, int out_size, void* d_ws, size_t ws_size,
                              hipStream_t stream) {
  (void)in_sizes; (void)n_in; (void)out_size;
  const float* inp   = (const float*)d_in[0];
  const int*   batch = (const int*)d_in[1];
  const float* gam   = (const float*)d_in[2];
  const float* bet   = (const float*)d_in[3];
  const float* mlp_w = (const float*)d_in[4];
  const float* mlp_b = (const float*)d_in[5];
  const float* w1    = (const float*)d_in[6];
  const float* w2    = (const float*)d_in[7];
  float* out = (float*)d_out;

  char* ws = (char*)d_ws;
  size_t o = 0;
  auto alloc = [&](size_t bytes) -> char* {
    char* r = ws + o;
    o = (o + bytes + 255) & ~(size_t)255;
    return r;
  };
  int*    starts = (int*)alloc(BG * 4);
  int*    ends   = (int*)alloc(BG * 4);
  int*    pos    = (int*)alloc((size_t)TT * 4);
  float*  x0     = (float*)alloc((size_t)TT * DD * 4);   // x0; y1 written in place -> x1
  half_t* x0t    = (half_t*)alloc((size_t)TT * DD * 2);  // x0^T f16; later reused as w1g
  half_t* xpeh   = (half_t*)alloc((size_t)TT * DD * 2);  // Xpe f16; later reused as w2g^T
  half_t* wh     = (half_t*)alloc((size_t)TT * DD * 2);  // W row-major f16
  half_t* wht    = (half_t*)alloc((size_t)TT * DD * 2);  // W^T f16; later reused as x2^T
  half_t* mwh    = (half_t*)alloc((size_t)DD * DD * 2);
  half_t* hth    = (half_t*)alloc((size_t)BG * DD * DD * 2);
  half_t* h2th   = (half_t*)alloc((size_t)BG * DD * DD * 2);
  if (o > ws_size) return;  // insufficient workspace

  half_t* w1g  = x0t;   // dead after h-GEMM
  half_t* w2gt = xpeh;  // dead after W-GEMM
  half_t* x2t  = wht;   // dead after h-GEMM

  k_init_seg<<<1, 64, 0, stream>>>(starts, ends);
  k_seg<<<(TT + 255) / 256, 256, 0, stream>>>(batch, starts, ends);
  k_pos<<<(TT + 255) / 256, 256, 0, stream>>>(batch, starts, pos);
  k_ln1<<<TT / 8, 256, 0, stream>>>(inp, gam, bet, pos, x0, x0t, xpeh);
  k_h16<<<(DD * DD + 255) / 256, 256, 0, stream>>>(mlp_w, mwh, DD * DD);
  // W = gelu(Xpe @ mlp_w^T + b)
  k_gemm_w<<<(TT + 31) / 32, 256, 0, stream>>>(xpeh, mwh, mlp_b, wh, wht);
  // h = gelu(W^T @ x0), per graph
  k_gemm_outer<<<BG * 8, 256, 0, stream>>>(wht, x0t, starts, ends, hth);
  // x1 = W @ h + x0 (in place)
  k_gemm_row<<<BG * 63, 256, 0, stream>>>(wh, hth, starts, ends, x0, x0);
  // x2 = LN(x1); stage x2^T, w2g^T, w1g
  k_ln2<<<TT / 8, 256, 0, stream>>>(x0, gam, bet, pos, w1, w2, x2t, w2gt, w1g);
  // h2 = gelu(w2g^T @ x2), per graph
  k_gemm_outer<<<BG * 8, 256, 0, stream>>>(w2gt, x2t, starts, ends, h2th);
  // out = w1g @ h2 + x1
  k_gemm_row<<<BG * 63, 256, 0, stream>>>(w1g, h2th, starts, ends, x0, out);
}
